// pointnet2_seg_ssg_32349693674191
// MI455X (gfx1250) — matrix operand order repaired and checked by reading, not compiled, not run on hardware
//
#include <hip/hip_runtime.h>

#define CDIV(a, b) (((a) + (b) - 1) / (b))

typedef __attribute__((ext_vector_type(16))) __bf16        v16bf;
typedef __attribute__((ext_vector_type(8)))  float         v8f;
typedef __attribute__((ext_vector_type(8)))  unsigned int  v8u;

// ======================================================================
// Operand packing into WMMA-native bf16 tile layouts (zero-padded K).
//   Apack: [r][ch][32] bf16, chunk K-order [0..7,16..23 | 8..15,24..31]
//          -> lane-half h loads 32 contiguous bytes at ((r*nch+ch)*2+h)
//   Wpack: [ch][col][32] bf16, same K-order as Apack -> half h = bytes h*32..h*32+31
// ======================================================================
__global__ __launch_bounds__(256) void pack_a_kernel(
    const float* __restrict__ A, unsigned short* __restrict__ dst,
    long long R, int K, int nch)
{
    long long total = R * (long long)nch * 32;
    long long t = (long long)blockIdx.x * blockDim.x + threadIdx.x;
    if (t >= total) return;
    int j = (int)(t & 31);
    long long rc = t >> 5;
    int ch = (int)(rc % nch);
    long long r = rc / nch;
    int ko = (j < 8) ? j : (j < 16) ? j + 8 : (j < 24) ? j - 8 : j;
    int k = ch * 32 + ko;
    float v = (k < K) ? A[r * K + k] : 0.0f;
    dst[t] = __builtin_bit_cast(unsigned short, (__bf16)v);
}

__global__ __launch_bounds__(256) void pack_w_kernel(
    const float* __restrict__ W, unsigned short* __restrict__ dst,
    int K, int N, int nch)
{
    long long total = (long long)nch * N * 32;
    long long t = (long long)blockIdx.x * blockDim.x + threadIdx.x;
    if (t >= total) return;
    int j = (int)(t & 31);
    long long cc = t >> 5;
    int col = (int)(cc % N);
    int ch = (int)(cc / N);
    int k = ch * 32 + ((j < 8) ? j : (j < 16) ? j + 8 : (j < 24) ? j - 8 : j);
    float v = (k < K) ? W[(size_t)k * N + col] : 0.0f;
    dst[t] = __builtin_bit_cast(unsigned short, (__bf16)v);
}

// ======================================================================
// WMMA bf16 GEMM on packed operands:  C[M,N] = A @ W (+bias)(+relu)
// One wave32 per 32x64 C macro-tile: 8 WMMAs/chunk, A reused 4x, B 2x.
// Software-pipelined: chunk ch+1 fragments are loaded before chunk ch's
// WMMAs execute, so s_wait_loadcnt is covered by 8 in-flight WMMAs.
// ======================================================================
__global__ __launch_bounds__(256) void gemm_wmma_kernel(
    const unsigned short* __restrict__ Apack, const unsigned short* __restrict__ Wpack,
    const float* __restrict__ bias, float* __restrict__ C,
    int M, int nch, int N, int do_relu)
{
    int wave = threadIdx.x >> 5;
    int lane = threadIdx.x & 31;
    int tilesM2 = (M + 31) >> 5;
    int tilesN4 = (N + 63) >> 6;
    int tile = blockIdx.x * 8 + wave;            // wave-uniform
    if (tile >= tilesM2 * tilesN4) return;       // uniform exit: EXEC stays full
    int tm = tile / tilesN4, tn4 = tile % tilesN4;
    int half = lane >> 4, l16 = lane & 15;

    // clamped addresses: OOB lanes read valid memory; their D rows/cols are never stored
    int rowA0 = tm * 32 + l16;      if (rowA0 >= M) rowA0 = M - 1;
    int rowA1 = tm * 32 + 16 + l16; if (rowA1 >= M) rowA1 = M - 1;
    int colBase = tn4 * 64 + l16;
    int colC0 = colBase, colC1 = colBase + 16, colC2 = colBase + 32, colC3 = colBase + 48;
    int cb0 = colC0 < N ? colC0 : N - 1;
    int cb1 = colC1 < N ? colC1 : N - 1;
    int cb2 = colC2 < N ? colC2 : N - 1;
    int cb3 = colC3 < N ? colC3 : N - 1;

    const v8u* Ap = (const v8u*)Apack;
    const v8u* Bp = (const v8u*)Wpack;
    size_t aIdx0 = ((size_t)rowA0 * nch) * 2 + half;
    size_t aIdx1 = ((size_t)rowA1 * nch) * 2 + half;

    v8f acc00 = {}, acc01 = {}, acc02 = {}, acc03 = {};
    v8f acc10 = {}, acc11 = {}, acc12 = {}, acc13 = {};

    // prologue: load chunk 0 fragments
    v8u au0 = Ap[aIdx0];
    v8u au1 = Ap[aIdx1];
    size_t bBase0 = (size_t)half;
    v8u bu0 = Bp[bBase0 + (size_t)cb0 * 2];
    v8u bu1 = Bp[bBase0 + (size_t)cb1 * 2];
    v8u bu2 = Bp[bBase0 + (size_t)cb2 * 2];
    v8u bu3 = Bp[bBase0 + (size_t)cb3 * 2];

    for (int ch = 0; ch < nch; ++ch) {
        // issue next chunk's loads before this chunk's WMMAs (pipelining)
        v8u au0n = au0, au1n = au1, bu0n = bu0, bu1n = bu1, bu2n = bu2, bu3n = bu3;
        if (ch + 1 < nch) {
            au0n = Ap[aIdx0 + (size_t)(ch + 1) * 2];
            au1n = Ap[aIdx1 + (size_t)(ch + 1) * 2];
            size_t bB = ((size_t)(ch + 1) * N) * 2 + half;
            bu0n = Bp[bB + (size_t)cb0 * 2];
            bu1n = Bp[bB + (size_t)cb1 * 2];
            bu2n = Bp[bB + (size_t)cb2 * 2];
            bu3n = Bp[bB + (size_t)cb3 * 2];
        }
        v16bf a0 = __builtin_bit_cast(v16bf, au0);
        v16bf a1 = __builtin_bit_cast(v16bf, au1);
        v16bf b0 = __builtin_bit_cast(v16bf, bu0);
        v16bf b1 = __builtin_bit_cast(v16bf, bu1);
        v16bf b2 = __builtin_bit_cast(v16bf, bu2);
        v16bf b3 = __builtin_bit_cast(v16bf, bu3);
        acc00 = __builtin_amdgcn_wmma_f32_16x16x32_bf16(false, a0, false, b0, (short)0, acc00, false, false);
        acc01 = __builtin_amdgcn_wmma_f32_16x16x32_bf16(false, a0, false, b1, (short)0, acc01, false, false);
        acc02 = __builtin_amdgcn_wmma_f32_16x16x32_bf16(false, a0, false, b2, (short)0, acc02, false, false);
        acc03 = __builtin_amdgcn_wmma_f32_16x16x32_bf16(false, a0, false, b3, (short)0, acc03, false, false);
        acc10 = __builtin_amdgcn_wmma_f32_16x16x32_bf16(false, a1, false, b0, (short)0, acc10, false, false);
        acc11 = __builtin_amdgcn_wmma_f32_16x16x32_bf16(false, a1, false, b1, (short)0, acc11, false, false);
        acc12 = __builtin_amdgcn_wmma_f32_16x16x32_bf16(false, a1, false, b2, (short)0, acc12, false, false);
        acc13 = __builtin_amdgcn_wmma_f32_16x16x32_bf16(false, a1, false, b3, (short)0, acc13, false, false);
        au0 = au0n; au1 = au1n;
        bu0 = bu0n; bu1 = bu1n; bu2 = bu2n; bu3 = bu3n;
    }

    // D layout: vgpr v holds row (tile_row_base + v + half*8)
#pragma unroll
    for (int mi = 0; mi < 2; ++mi) {
        v8f* accs0 = mi == 0 ? &acc00 : &acc10;
#pragma unroll
        for (int v = 0; v < 8; ++v) {
            int r = tm * 32 + mi * 16 + v + half * 8;
            if (r >= M) continue;
            float* Crow = C + (size_t)r * N;
            if (colC0 < N) {
                float val = accs0[0][v];
                if (bias) val += bias[colC0];
                if (do_relu) val = fmaxf(val, 0.0f);
                Crow[colC0] = val;
            }
            if (colC1 < N) {
                float val = accs0[1][v];
                if (bias) val += bias[colC1];
                if (do_relu) val = fmaxf(val, 0.0f);
                Crow[colC1] = val;
            }
            if (colC2 < N) {
                float val = accs0[2][v];
                if (bias) val += bias[colC2];
                if (do_relu) val = fmaxf(val, 0.0f);
                Crow[colC2] = val;
            }
            if (colC3 < N) {
                float val = accs0[3][v];
                if (bias) val += bias[colC3];
                if (do_relu) val = fmaxf(val, 0.0f);
                Crow[colC3] = val;
            }
        }
    }
}

// ======================================================================
// Column statistics (batchnorm over all leading axes): mean/var per column
// ======================================================================
__global__ __launch_bounds__(256) void col_stats_kernel(
    const float* __restrict__ x, float* __restrict__ mean, float* __restrict__ var,
    long long R, int C)
{
    int c = blockIdx.x;
    float s = 0.f, s2 = 0.f;
    for (long long r = threadIdx.x; r < R; r += blockDim.x) {
        float v = x[r * C + c];
        s += v; s2 += v * v;
    }
    __shared__ float sh[256], sh2[256];
    sh[threadIdx.x] = s; sh2[threadIdx.x] = s2;
    __syncthreads();
    for (int st = 128; st > 0; st >>= 1) {
        if ((int)threadIdx.x < st) { sh[threadIdx.x] += sh[threadIdx.x + st]; sh2[threadIdx.x] += sh2[threadIdx.x + st]; }
        __syncthreads();
    }
    if (threadIdx.x == 0) {
        float m = sh[0] / (float)R;
        mean[c] = m;
        var[c] = sh2[0] / (float)R - m * m;
    }
}

__global__ __launch_bounds__(256) void bn_relu_kernel(
    float* __restrict__ x, const float* __restrict__ mean, const float* __restrict__ var,
    const float* __restrict__ g, const float* __restrict__ b, long long total, int C)
{
    long long t = (long long)blockIdx.x * blockDim.x + threadIdx.x;
    if (t >= total) return;
    int c = (int)(t % C);
    float v = x[t];
    v = g[c] * (v - mean[c]) * rsqrtf(var[c] + 1e-5f) + b[c];
    x[t] = fmaxf(v, 0.0f);
}

// ======================================================================
// Farthest point sampling: one block per batch, LDS-resident distance array
// ======================================================================
__global__ __launch_bounds__(256) void fps_kernel(
    const float* __restrict__ xyz, int Npts, int M, int* __restrict__ out_idx)
{
    int b = blockIdx.x;
    const float* X = xyz + (size_t)b * Npts * 3;
    __shared__ float dist[4096];
    __shared__ float rmax[256];
    __shared__ int   rarg[256];
    __shared__ int   far_s;
    for (int i = threadIdx.x; i < Npts; i += blockDim.x) dist[i] = 1e10f;
    if (threadIdx.x == 0) far_s = 0;
    __syncthreads();
    for (int m = 0; m < M; ++m) {
        int far = far_s;
        if (threadIdx.x == 0) out_idx[b * M + m] = far;
        float cx = X[far * 3 + 0], cy = X[far * 3 + 1], cz = X[far * 3 + 2];
        float best = -1.0f; int bestI = 0x7fffffff;
        for (int i = threadIdx.x; i < Npts; i += blockDim.x) {
            float dx = X[i * 3 + 0] - cx, dy = X[i * 3 + 1] - cy, dz = X[i * 3 + 2] - cz;
            float d = dx * dx + dy * dy + dz * dz;
            float nd = fminf(dist[i], d);
            dist[i] = nd;
            if (nd > best || (nd == best && i < bestI)) { best = nd; bestI = i; }
        }
        rmax[threadIdx.x] = best; rarg[threadIdx.x] = bestI;
        __syncthreads();
        for (int s = 128; s > 0; s >>= 1) {
            if ((int)threadIdx.x < s) {
                float om = rmax[threadIdx.x + s]; int oi = rarg[threadIdx.x + s];
                if (om > rmax[threadIdx.x] || (om == rmax[threadIdx.x] && oi < rarg[threadIdx.x])) {
                    rmax[threadIdx.x] = om; rarg[threadIdx.x] = oi;
                }
            }
            __syncthreads();
        }
        if (threadIdx.x == 0) far_s = rarg[0];
        __syncthreads();
    }
}

__global__ __launch_bounds__(256) void gather_xyz_kernel(
    const float* __restrict__ xyz, const int* __restrict__ fidx,
    float* __restrict__ out, int Bv, int Npts, int M)
{
    int t = blockIdx.x * blockDim.x + threadIdx.x;
    if (t >= Bv * M) return;
    int b = t / M;
    int pi = fidx[t];
    const float* p = xyz + ((size_t)b * Npts + pi) * 3;
    out[t * 3 + 0] = p[0]; out[t * 3 + 1] = p[1]; out[t * 3 + 2] = p[2];
}

// ======================================================================
// Ball query: first-K in-range indices in increasing order, padded w/ first
// ======================================================================
__global__ __launch_bounds__(256) void ball_query_kernel(
    const float* __restrict__ xyz, const float* __restrict__ new_xyz,
    int* __restrict__ nidx, int Bv, int Npts, int M, int Kn, float r2)
{
    int t = blockIdx.x * blockDim.x + threadIdx.x;
    if (t >= Bv * M) return;
    int b = t / M;
    const float* X = xyz + (size_t)b * Npts * 3;
    float qx = new_xyz[t * 3 + 0], qy = new_xyz[t * 3 + 1], qz = new_xyz[t * 3 + 2];
    int* out = nidx + (size_t)t * Kn;
    int cnt = 0, first = 0;
    for (int i = 0; i < Npts && cnt < Kn; ++i) {
        float dx = X[i * 3 + 0] - qx, dy = X[i * 3 + 1] - qy, dz = X[i * 3 + 2] - qz;
        if (dx * dx + dy * dy + dz * dz <= r2) {
            if (cnt == 0) first = i;
            out[cnt++] = i;
        }
    }
    for (; cnt < Kn; ++cnt) out[cnt] = first;
}

// Build grouped feature rows: [ (xyz[nidx]-new_xyz) | pts[nidx] ]
__global__ __launch_bounds__(256) void group_concat_kernel(
    const float* __restrict__ xyz, const float* __restrict__ pts,
    const float* __restrict__ new_xyz, const int* __restrict__ nidx,
    float* __restrict__ A, int Bv, int Npts, int M, int Kn, int Cp)
{
    int Cin = 3 + Cp;
    long long total = (long long)Bv * M * Kn * Cin;
    long long t = (long long)blockIdx.x * blockDim.x + threadIdx.x;
    if (t >= total) return;
    int c = (int)(t % Cin);
    long long r = t / Cin;                 // (b*M + m)*Kn + k
    long long bm = r / Kn;
    int b = (int)(bm / M), m = (int)(bm % M);
    int pi = nidx[r];
    if (c < 3)
        A[t] = xyz[((size_t)b * Npts + pi) * 3 + c] - new_xyz[((size_t)b * M + m) * 3 + c];
    else
        A[t] = pts[((size_t)b * Npts + pi) * Cp + (c - 3)];
}

__global__ __launch_bounds__(256) void max_pool_kernel(
    const float* __restrict__ feat, float* __restrict__ out,
    long long BM, int Kn, int C)
{
    long long t = (long long)blockIdx.x * blockDim.x + threadIdx.x;
    if (t >= BM * C) return;
    int c = (int)(t % C);
    long long r = t / C;
    const float* F = feat + r * (size_t)Kn * C + c;
    float m = F[0];
    for (int k = 1; k < Kn; ++k) m = fmaxf(m, F[(size_t)k * C]);
    out[t] = m;
}

__global__ __launch_bounds__(256) void mean_rows_kernel(
    const float* __restrict__ x, float* __restrict__ out, int Bv, int Mr, int C)
{
    int t = blockIdx.x * blockDim.x + threadIdx.x;
    if (t >= Bv * C) return;
    int b = t / C, c = t % C;
    const float* X = x + (size_t)b * Mr * C + c;
    float s = 0.f;
    for (int m = 0; m < Mr; ++m) s += X[(size_t)m * C];
    out[t] = s / (float)Mr;
}

__global__ __launch_bounds__(256) void copy_cols_kernel(
    const float* __restrict__ src, float* __restrict__ dst,
    long long R, int Cs, int Cd, int coff)
{
    long long t = (long long)blockIdx.x * blockDim.x + threadIdx.x;
    if (t >= R * Cs) return;
    int c = (int)(t % Cs);
    long long r = t / Cs;
    dst[r * Cd + coff + c] = src[t];
}

// broadcast per-batch feature vector over Mrep rows (fp with single source point)
__global__ __launch_bounds__(256) void broadcast_rows_kernel(
    const float* __restrict__ src, float* __restrict__ dst,
    int Bv, int Mrep, int C, int Cd, int coff)
{
    long long total = (long long)Bv * Mrep * C;
    long long t = (long long)blockIdx.x * blockDim.x + threadIdx.x;
    if (t >= total) return;
    int c = (int)(t % C);
    long long r = t / C;
    int b = (int)(r / Mrep);
    dst[r * Cd + coff + c] = src[(size_t)b * C + c];
}

// 3-NN indices + inverse-distance weights
__global__ __launch_bounds__(256) void top3_kernel(
    const float* __restrict__ xyz1, const float* __restrict__ xyz2,
    int* __restrict__ idx3, float* __restrict__ w3, int Bv, int N1, int M2)
{
    int t = blockIdx.x * blockDim.x + threadIdx.x;
    if (t >= Bv * N1) return;
    int b = t / N1;
    const float* X2 = xyz2 + (size_t)b * M2 * 3;
    float px = xyz1[(size_t)t * 3 + 0], py = xyz1[(size_t)t * 3 + 1], pz = xyz1[(size_t)t * 3 + 2];
    float d0 = 1e30f, d1 = 1e30f, d2 = 1e30f; int i0 = 0, i1 = 0, i2 = 0;
    for (int j = 0; j < M2; ++j) {
        float dx = X2[j * 3 + 0] - px, dy = X2[j * 3 + 1] - py, dz = X2[j * 3 + 2] - pz;
        float d = dx * dx + dy * dy + dz * dz;
        if (d < d0)      { d2 = d1; i2 = i1; d1 = d0; i1 = i0; d0 = d; i0 = j; }
        else if (d < d1) { d2 = d1; i2 = i1; d1 = d;  i1 = j; }
        else if (d < d2) { d2 = d;  i2 = j; }
    }
    float w0 = 1.f / (fmaxf(d0, 0.f) + 1e-8f);
    float w1 = 1.f / (fmaxf(d1, 0.f) + 1e-8f);
    float w2 = 1.f / (fmaxf(d2, 0.f) + 1e-8f);
    float ws = w0 + w1 + w2;
    idx3[t * 3 + 0] = i0; idx3[t * 3 + 1] = i1; idx3[t * 3 + 2] = i2;
    w3[t * 3 + 0] = w0 / ws; w3[t * 3 + 1] = w1 / ws; w3[t * 3 + 2] = w2 / ws;
}

__global__ __launch_bounds__(256) void interp_gather_kernel(
    const float* __restrict__ pts2, const int* __restrict__ idx3,
    const float* __restrict__ w3, float* __restrict__ dst,
    int Bv, int N1, int M2, int C2, int Cin, int coff)
{
    long long total = (long long)Bv * N1 * C2;
    long long t = (long long)blockIdx.x * blockDim.x + threadIdx.x;
    if (t >= total) return;
    int c = (int)(t % C2);
    long long r = t / C2;                  // b*N1 + n
    int b = (int)(r / N1);
    const int* id = idx3 + r * 3;
    const float* w = w3 + r * 3;
    const float* P = pts2 + (size_t)b * M2 * C2;
    float v = w[0] * P[(size_t)id[0] * C2 + c]
            + w[1] * P[(size_t)id[1] * C2 + c]
            + w[2] * P[(size_t)id[2] * C2 + c];
    dst[r * Cin + coff + c] = v;
}

// x = 0.5*relu(t) + 0.5*x
__global__ __launch_bounds__(256) void adapter_combine_kernel(
    float* __restrict__ x, const float* __restrict__ t, long long n)
{
    long long i = (long long)blockIdx.x * blockDim.x + threadIdx.x;
    if (i >= n) return;
    x[i] = 0.5f * fmaxf(t[i], 0.0f) + 0.5f * x[i];
}

// cosine hinge loss, one block per row, atomicAdd mean into loss scalar
__global__ __launch_bounds__(128) void cos_loss_kernel(
    const float* __restrict__ a, const float* __restrict__ b,
    float* __restrict__ loss, int Bv, int D, float thr)
{
    int row = blockIdx.x;
    float aa = 0.f, bb = 0.f, ab = 0.f;
    for (int i = threadIdx.x; i < D; i += blockDim.x) {
        float x = a[(size_t)row * D + i], y = b[(size_t)row * D + i];
        aa += x * x; bb += y * y; ab += x * y;
    }
    __shared__ float saa[128], sbb[128], sab[128];
    saa[threadIdx.x] = aa; sbb[threadIdx.x] = bb; sab[threadIdx.x] = ab;
    __syncthreads();
    for (int s = 64; s > 0; s >>= 1) {
        if ((int)threadIdx.x < s) {
            saa[threadIdx.x] += saa[threadIdx.x + s];
            sbb[threadIdx.x] += sbb[threadIdx.x + s];
            sab[threadIdx.x] += sab[threadIdx.x + s];
        }
        __syncthreads();
    }
    if (threadIdx.x == 0) {
        float na = fmaxf(sqrtf(saa[0]), 1e-8f);
        float nb = fmaxf(sqrtf(sbb[0]), 1e-8f);
        float c = sab[0] / (na * nb);
        atomicAdd(loss, fmaxf(thr - c, 0.0f) / (float)Bv);
    }
}

__device__ __constant__ int SEG_TABLE_D[50] = {
    0,0,0,0,1,1,2,2,3,3,3,3,4,4,4,4,5,5,5,6,6,6,7,7,8,
    8,8,8,9,9,10,10,10,10,10,10,11,11,12,12,12,13,13,13,14,14,14,15,15,15};

__global__ void ce_kernel(const float* __restrict__ logits, const int* __restrict__ label,
                          float* __restrict__ loss, int Bv, int Ncl)
{
    int b = threadIdx.x;
    if (b >= Bv) return;
    int lbl = SEG_TABLE_D[label[b]];
    const float* L = logits + (size_t)b * Ncl;
    float mx = -1e30f;
    for (int i = 0; i < Ncl; ++i) mx = fmaxf(mx, L[i]);
    float s = 0.f;
    for (int i = 0; i < Ncl; ++i) s += __expf(L[i] - mx);
    float lp = L[lbl] - mx - __logf(s);
    atomicAdd(loss, -lp / (float)Bv);
}

// (B*Np, Ncl) -> d_out[1 + (b*Ncl + c)*Np + n]
__global__ __launch_bounds__(256) void transpose_out_kernel(
    const float* __restrict__ net, float* __restrict__ out, int Bv, int Np, int Ncl)
{
    long long total = (long long)Bv * Np * Ncl;
    long long t = (long long)blockIdx.x * blockDim.x + threadIdx.x;
    if (t >= total) return;
    int c = (int)(t % Ncl);
    long long r = t / Ncl;
    int n = (int)(r % Np), b = (int)(r / Np);
    out[1 + ((size_t)b * Ncl + c) * Np + n] = net[t];
}

__global__ void zero_loss_kernel(float* out) { out[0] = 0.0f; }

// ======================================================================
// Host-side orchestration
// ======================================================================
extern "C" void kernel_launch(void* const* d_in, const int* in_sizes, int n_in,
                              void* d_out, int out_size, void* d_ws, size_t ws_size,
                              hipStream_t stream)
{
    (void)in_sizes; (void)n_in; (void)out_size; (void)ws_size;
    const int B = 16, N0 = 4096, M1 = 512, K1 = 32, M2 = 128, K2 = 64, NCLS = 50;

    const float* l0_xyz    = (const float*)d_in[0];
    const float* l0_points = (const float*)d_in[1];
    const int*   label     = (const int*)  d_in[2];
    auto F = [&](int i) { return (const float*)d_in[i]; };
    // params in setup_inputs() insertion order (lists in order, tuples W,g,b):
    // sa1:3..11  sa2:12..20  sa3:21..29  fp1:30..35  fp2:36..41  fp3:42..50
    // conv1_w:51 conv1_g:52 conv1_b:53  cls_w:54 cls_bias:55
    // fcA:56..58 fcB:59..61 fc2:62..64  clsfc:65..67 adapter_w:68 clip_w1:69 clip_w2:70

    float* out = (float*)d_out;

    // ---- workspace allocator ----
    char* ws = (char*)d_ws;
    size_t cur = 0;
    auto alloc = [&](size_t bytes) -> void* {
        cur = (cur + 255) & ~(size_t)255;
        void* p = ws + cur;
        cur += bytes;
        return p;
    };
    const size_t PPBYTES = (size_t)262144 * 128 * sizeof(float);   // largest f32 intermediate
    float* PP0 = (float*)alloc(PPBYTES);
    float* PP1 = (float*)alloc(PPBYTES);
    unsigned short* PKA = (unsigned short*)alloc((size_t)48 * 1024 * 1024); // packed A (bf16)
    unsigned short* PKW = (unsigned short*)alloc((size_t)4  * 1024 * 1024); // packed W (bf16)
    float* feat1 = (float*)alloc((size_t)B * 512 * 4);
    float* feat2 = (float*)alloc((size_t)B * 512 * 4);
    float* feat3 = (float*)alloc((size_t)B * 512 * 4);
    float* logitsb = (float*)alloc((size_t)B * 16 * 4);
    float* meanb = (float*)alloc((size_t)B * 3 * 4);
    float* statm = (float*)alloc(1024 * 4);
    float* statv = (float*)alloc(1024 * 4);
    int*   fidx1 = (int*)alloc((size_t)B * M1 * 4);
    float* l1_xyz = (float*)alloc((size_t)B * M1 * 3 * 4);
    int*   nidx1 = (int*)alloc((size_t)B * M1 * K1 * 4);
    float* l1_points = (float*)alloc((size_t)B * M1 * 128 * 4);
    int*   fidx2 = (int*)alloc((size_t)B * M2 * 4);
    float* l2_xyz = (float*)alloc((size_t)B * M2 * 3 * 4);
    int*   nidx2 = (int*)alloc((size_t)B * M2 * K2 * 4);
    float* l2_points = (float*)alloc((size_t)B * M2 * 256 * 4);
    float* l3_points = (float*)alloc((size_t)B * 1024 * 4);
    float* l2p = (float*)alloc((size_t)B * M2 * 256 * 4);
    float* l1p = (float*)alloc((size_t)B * M1 * 128 * 4);
    float* l0p = (float*)alloc((size_t)B * N0 * 128 * 4);
    int*   idx3 = (int*)alloc((size_t)B * N0 * 3 * 4);
    float* w3   = (float*)alloc((size_t)B * N0 * 3 * 4);
    float* pf0  = (float*)alloc((size_t)B * 512 * 4);
    float* pfT  = (float*)alloc((size_t)B * 512 * 4);
    float* net_tmp = (float*)alloc((size_t)B * N0 * NCLS * 4);

    auto EW = [&](long long n) { return (int)CDIV(n, 256); };

    // pack A + pack W + WMMA GEMM
    auto run_gemm = [&](const float* A, const float* Wt, const float* bias, float* C,
                        long long R, int K, int N, int relu) {
        int nch = CDIV(K, 32);
        pack_a_kernel<<<EW(R * nch * 32), 256, 0, stream>>>(A, PKA, R, K, nch);
        pack_w_kernel<<<EW((long long)nch * N * 32), 256, 0, stream>>>(Wt, PKW, K, N, nch);
        int tiles = CDIV(R, 32) * CDIV(N, 64);
        gemm_wmma_kernel<<<CDIV(tiles, 8), 256, 0, stream>>>(PKA, PKW, bias, C, (int)R, nch, N, relu);
    };
    // GEMM -> global BN stats -> BN+ReLU
    auto mlp_layer = [&](const float* A, const float* Wt, const float* g, const float* bb,
                         float* Cout, long long R, int K, int Nc) {
        run_gemm(A, Wt, nullptr, Cout, R, K, Nc, 0);
        col_stats_kernel<<<Nc, 256, 0, stream>>>(Cout, statm, statv, R, Nc);
        bn_relu_kernel<<<EW(R * Nc), 256, 0, stream>>>(Cout, statm, statv, g, bb, R * Nc, Nc);
    };
    // adapter: x = 0.5*relu(x @ adapter_w) + 0.5*x
    auto adapter = [&](float* x, float* tmp, int Rr, int D) {
        run_gemm(x, F(68), nullptr, tmp, Rr, D, D, 0);
        adapter_combine_kernel<<<EW((long long)Rr * D), 256, 0, stream>>>(x, tmp, (long long)Rr * D);
    };
    // clip_model: feat = relu(mean(xyz) @ clip_w1); logits = feat @ clip_w2; ce
    auto clip = [&](const float* xyz, int Np, float* featbuf) {
        mean_rows_kernel<<<EW(B * 3), 256, 0, stream>>>(xyz, meanb, B, Np, 3);
        run_gemm(meanb, F(69), nullptr, featbuf, B, 3, 512, 1);
        run_gemm(featbuf, F(70), nullptr, logitsb, B, 512, 16, 0);
        ce_kernel<<<1, 32, 0, stream>>>(logitsb, label, out, B, 16);
    };

    zero_loss_kernel<<<1, 1, 0, stream>>>(out);

    // ---------------- clip #1 ----------------
    clip(l0_xyz, N0, feat1);

    // ---------------- SA1: 4096 -> 512, K=32, mlp 6->64->64->128 ----------------
    fps_kernel<<<B, 256, 0, stream>>>(l0_xyz, N0, M1, fidx1);
    gather_xyz_kernel<<<EW(B * M1), 256, 0, stream>>>(l0_xyz, fidx1, l1_xyz, B, N0, M1);
    ball_query_kernel<<<EW(B * M1), 256, 0, stream>>>(l0_xyz, l1_xyz, nidx1, B, N0, M1, K1, 0.04f);
    {
        long long R = (long long)B * M1 * K1; // 262144
        group_concat_kernel<<<EW(R * 6), 256, 0, stream>>>(l0_xyz, l0_points, l1_xyz, nidx1, PP0, B, N0, M1, K1, 3);
        mlp_layer(PP0, F(3), F(4), F(5),   PP1, R,   6,  64);
        mlp_layer(PP1, F(6), F(7), F(8),   PP0, R,  64,  64);
        mlp_layer(PP0, F(9), F(10), F(11), PP1, R,  64, 128);
        max_pool_kernel<<<EW((long long)B * M1 * 128), 256, 0, stream>>>(PP1, l1_points, (long long)B * M1, K1, 128);
    }
    // pf1 path: mean -> fcA -> fcB -> adapter -> cos(feat1, ., 0.8)
    mean_rows_kernel<<<EW(B * 128), 256, 0, stream>>>(l1_points, pf0, B, M1, 128);
    mlp_layer(pf0, F(56), F(57), F(58), pfT, B, 128, 256);
    mlp_layer(pfT, F(59), F(60), F(61), pf0, B, 256, 512);
    adapter(pf0, pfT, B, 512);
    cos_loss_kernel<<<B, 128, 0, stream>>>(feat1, pf0, out, B, 512, 0.8f);

    // ---------------- clip #2 ----------------
    clip(l1_xyz, M1, feat2);

    // ---------------- SA2: 512 -> 128, K=64, mlp 131->128->128->256 ----------------
    fps_kernel<<<B, 256, 0, stream>>>(l1_xyz, M1, M2, fidx2);
    gather_xyz_kernel<<<EW(B * M2), 256, 0, stream>>>(l1_xyz, fidx2, l2_xyz, B, M1, M2);
    ball_query_kernel<<<EW(B * M2), 256, 0, stream>>>(l1_xyz, l2_xyz, nidx2, B, M1, M2, K2, 0.16f);
    {
        long long R = (long long)B * M2 * K2; // 131072
        group_concat_kernel<<<EW(R * 131), 256, 0, stream>>>(l1_xyz, l1_points, l2_xyz, nidx2, PP0, B, M1, M2, K2, 128);
        mlp_layer(PP0, F(12), F(13), F(14), PP1, R, 131, 128);
        mlp_layer(PP1, F(15), F(16), F(17), PP0, R, 128, 128);
        mlp_layer(PP0, F(18), F(19), F(20), PP1, R, 128, 256);
        max_pool_kernel<<<EW((long long)B * M2 * 256), 256, 0, stream>>>(PP1, l2_points, (long long)B * M2, K2, 256);
    }
    // pf2 path: mean -> fc2 -> adapter -> cos(feat2, ., 0.8)
    mean_rows_kernel<<<EW(B * 256), 256, 0, stream>>>(l2_points, pf0, B, M2, 256);
    mlp_layer(pf0, F(62), F(63), F(64), pfT, B, 256, 512);
    adapter(pfT, pf0, B, 512);
    cos_loss_kernel<<<B, 128, 0, stream>>>(feat2, pfT, out, B, 512, 0.8f);

    // ---------------- clip #3 ----------------
    clip(l2_xyz, M2, feat3);

    // ---------------- SA3 (group-all): mlp 259->256->512->1024, max over 128 ----------------
    {
        long long R = (long long)B * M2; // 2048
        copy_cols_kernel<<<EW(R * 3), 256, 0, stream>>>(l2_xyz, PP0, R, 3, 259, 0);
        copy_cols_kernel<<<EW(R * 256), 256, 0, stream>>>(l2_points, PP0, R, 256, 259, 3);
        mlp_layer(PP0, F(21), F(22), F(23), PP1, R, 259, 256);
        mlp_layer(PP1, F(24), F(25), F(26), PP0, R, 256, 512);
        mlp_layer(PP0, F(27), F(28), F(29), PP1, R, 512, 1024);
        max_pool_kernel<<<EW((long long)B * 1024), 256, 0, stream>>>(PP1, l3_points, (long long)B, M2, 1024);
    }
    // pf3 path: clsfc -> adapter -> cos(feat3, ., 0.8)
    mlp_layer(l3_points, F(65), F(66), F(67), pf0, B, 1024, 512);
    adapter(pf0, pfT, B, 512);
    cos_loss_kernel<<<B, 128, 0, stream>>>(feat3, pf0, out, B, 512, 0.8f);

    // ---------------- FP1: broadcast l3 -> concat(l2_points) -> mlp 1280->256->256 ----------------
    {
        long long R = (long long)B * M2; // 2048
        copy_cols_kernel<<<EW(R * 256), 256, 0, stream>>>(l2_points, PP0, R, 256, 1280, 0);
        broadcast_rows_kernel<<<EW(R * 1024), 256, 0, stream>>>(l3_points, PP0, B, M2, 1024, 1280, 256);
        mlp_layer(PP0, F(30), F(31), F(32), PP1, R, 1280, 256);
        mlp_layer(PP1, F(33), F(34), F(35), l2p, R, 256, 256);
    }
    // ---------------- FP2: 3-NN interp l2p -> concat(l1_points) -> mlp 384->256->128 ----------------
    {
        long long R = (long long)B * M1; // 8192
        top3_kernel<<<EW(B * M1), 256, 0, stream>>>(l1_xyz, l2_xyz, idx3, w3, B, M1, M2);
        copy_cols_kernel<<<EW(R * 128), 256, 0, stream>>>(l1_points, PP0, R, 128, 384, 0);
        interp_gather_kernel<<<EW(R * 256), 256, 0, stream>>>(l2p, idx3, w3, PP0, B, M1, M2, 256, 384, 128);
        mlp_layer(PP0, F(36), F(37), F(38), PP1, R, 384, 256);
        mlp_layer(PP1, F(39), F(40), F(41), l1p, R, 256, 128);
    }
    // ---------------- FP3: 3-NN interp l1p -> concat(l0_points,l0_xyz) -> mlp 134->128->128->128 ----
    {
        long long R = (long long)B * N0; // 65536
        top3_kernel<<<EW(B * N0), 256, 0, stream>>>(l0_xyz, l1_xyz, idx3, w3, B, N0, M1);
        copy_cols_kernel<<<EW(R * 3), 256, 0, stream>>>(l0_points, PP0, R, 3, 134, 0);
        copy_cols_kernel<<<EW(R * 3), 256, 0, stream>>>(l0_xyz, PP0, R, 3, 134, 3);
        interp_gather_kernel<<<EW(R * 128), 256, 0, stream>>>(l1p, idx3, w3, PP0, B, N0, M1, 128, 134, 6);
        mlp_layer(PP0, F(42), F(43), F(44), PP1, R, 134, 128);
        mlp_layer(PP1, F(45), F(46), F(47), PP0, R, 128, 128);
        mlp_layer(PP0, F(48), F(49), F(50), l0p, R, 128, 128);
    }
    // pf4 path: mean(l0p) -> fcA -> fcB -> adapter -> cos(feat1, ., 0.8)
    mean_rows_kernel<<<EW(B * 128), 256, 0, stream>>>(l0p, pf0, B, N0, 128);
    mlp_layer(pf0, F(56), F(57), F(58), pfT, B, 128, 256);
    mlp_layer(pfT, F(59), F(60), F(61), pf0, B, 256, 512);
    adapter(pf0, pfT, B, 512);
    cos_loss_kernel<<<B, 128, 0, stream>>>(feat1, pf0, out, B, 512, 0.8f);

    // ---------------- head: conv1 (BN+ReLU) -> cls (+bias) -> transpose ----------------
    {
        long long R = (long long)B * N0; // 65536
        mlp_layer(l0p, F(51), F(52), F(53), PP0, R, 128, 128);
        run_gemm(PP0, F(54), F(55), net_tmp, R, 128, NCLS, 0);
        transpose_out_kernel<<<EW(R * NCLS), 256, 0, stream>>>(net_tmp, out, B, N0, NCLS);
    }
}
